// CrossAttentionBlock_20907900797456
// MI455X (gfx1250) — matrix operand order repaired and checked by reading, not compiled, not run on hardware
//
#include <hip/hip_runtime.h>

// Sizes fixed by the reference problem.
#define N_ATOMS 262144
#define BATCH   4096
#define HID     256
#define SCALE   0.125f          // (HIDDEN/HEADS)^-0.5 = 64^-0.5
#define LN_EPS  1e-5f
#define SM_EPS  1e-16f

#define LDA 264                 // bf16 LDS row stride (pad vs bank conflicts)
#define LDF 260                 // f32  LDS row stride

typedef __attribute__((ext_vector_type(16))) __bf16 v16bf;
typedef __attribute__((ext_vector_type(8)))  float  v8f;

union BfPack { unsigned int u[8]; v16bf v; };

__device__ __forceinline__ unsigned short f2bf(float f) {
  unsigned int u = __float_as_uint(f);
  u += 0x7FFFu + ((u >> 16) & 1u);   // round-to-nearest-even
  return (unsigned short)(u >> 16);
}
// order-preserving float<->uint key for atomicMax on floats
__device__ __forceinline__ unsigned int fkey(float f) {
  unsigned int u = __float_as_uint(f);
  return (u & 0x80000000u) ? ~u : (u | 0x80000000u);
}
__device__ __forceinline__ float funkey(unsigned int k) {
  unsigned int u = (k & 0x80000000u) ? (k ^ 0x80000000u) : ~k;
  return __uint_as_float(u);
}

// ---------------------------------------------------------------------------
// Kernel 0: weights f32 -> bf16; zero segment max-keys and sums (every call).
// ---------------------------------------------------------------------------
__global__ void k_init(const float* __restrict__ Wq, const float* __restrict__ Wk,
                       const float* __restrict__ Wv,
                       unsigned short* __restrict__ wqbf, unsigned short* __restrict__ wkbf,
                       unsigned short* __restrict__ wvbf,
                       unsigned int* __restrict__ segkey, float* __restrict__ segsum) {
  int i = blockIdx.x * blockDim.x + threadIdx.x;
  if (i < HID * HID) {
    wqbf[i] = f2bf(Wq[i]);
    wkbf[i] = f2bf(Wk[i]);
    wvbf[i] = f2bf(Wv[i]);
  }
  if (i < BATCH) { segkey[i] = 0u; segsum[i] = 0.0f; }
}

// ---------------------------------------------------------------------------
// Kernel 1: per 16-row tile, WMMA Q = gather(prot)@Wq+bq, K = drug@Wk+bk in
// LDS, per-row dot(Q,K)*SCALE -> ws_dots, segment max via atomicMax on keys.
// ---------------------------------------------------------------------------
__global__ __launch_bounds__(256)
void k_qk_dots(const float* __restrict__ drug, const float* __restrict__ prot,
               const int* __restrict__ bidx,
               const unsigned short* __restrict__ wqbf, const unsigned short* __restrict__ wkbf,
               const float* __restrict__ bq, const float* __restrict__ bk,
               float* __restrict__ ws_dots, unsigned int* __restrict__ segkey) {
  __shared__ unsigned short sAp[16 * LDA];   // gathered protein rows, bf16
  __shared__ unsigned short sAd[16 * LDA];   // drug rows, bf16
  __shared__ float sQ[16 * LDF];
  __shared__ float sK[16 * LDF];
  __shared__ float sred[256];

  const int tid = threadIdx.x;
  const int rowbase = blockIdx.x * 16;
  const int r  = tid >> 4;          // row in tile this thread stages/reduces
  const int cb = (tid & 15) << 4;   // 16-col chunk base

  // Stage A tiles as bf16 (coalesced float4 loads, 16 floats / thread).
  {
    const float4* dp  = (const float4*)(drug + (size_t)rowbase * HID);
    const int idx     = bidx[rowbase + r];
    const float4* pp  = (const float4*)(prot + (size_t)idx * HID);
#pragma unroll
    for (int q = 0; q < 4; ++q) {
      float4 v = dp[tid * 4 + q];
      unsigned short* o = sAd + r * LDA + cb + q * 4;
      o[0] = f2bf(v.x); o[1] = f2bf(v.y); o[2] = f2bf(v.z); o[3] = f2bf(v.w);
      float4 w = pp[(tid & 15) * 4 + q];
      unsigned short* o2 = sAp + r * LDA + cb + q * 4;
      o2[0] = f2bf(w.x); o2[1] = f2bf(w.y); o2[2] = f2bf(w.z); o2[3] = f2bf(w.w);
    }
  }
  __syncthreads();

  const int lane = tid & 31;
  const int wv   = tid >> 5;        // wave id 0..7
  const int hh   = lane >> 4;       // lane half
  const int ml   = lane & 15;       // A row (M) == C column (N) index

  // 32 column-tiles total: t<16 -> Q, t>=16 -> K; 4 per wave.
#pragma unroll
  for (int i = 0; i < 4; ++i) {
    const int t    = wv * 4 + i;
    const int mat  = t >> 4;        // 0 = Q (protein), 1 = K (drug)
    const int ncol = t & 15;
    const unsigned short* Asrc = mat ? sAd : sAp;
    const unsigned short* Wsrc = mat ? wkbf : wqbf;
    const float bias = (mat ? bk : bq)[ncol * 16 + ml];
    v8f acc;
#pragma unroll
    for (int j = 0; j < 8; ++j) acc[j] = bias;
#pragma unroll
    for (int k0 = 0; k0 < HID; k0 += 32) {
      BfPack a, b;
      // A 16x32 bf16 layout: lanes0-15 M=lane,K=k0+{0..7,16..23}; lanes16-31 +8
      const unsigned short* ar = Asrc + ml * LDA + k0 + hh * 8;
      const unsigned int* p1 = (const unsigned int*)ar;
      const unsigned int* p2 = (const unsigned int*)(ar + 16);
#pragma unroll
      for (int q = 0; q < 4; ++q) { a.u[q] = p1[q]; a.u[4 + q] = p2[q]; }
      // B 32x16 bf16: lanes0-15 N=lane,K=k0+{0..7,16..23}; lanes16-31 +8
      const unsigned short* pb =
          Wsrc + (size_t)(k0 + hh * 8) * HID + ncol * 16 + ml;
#pragma unroll
      for (int q = 0; q < 4; ++q) {
        b.u[q] = (unsigned int)pb[(2 * q) * HID] | ((unsigned int)pb[(2 * q + 1) * HID] << 16);
        b.u[4 + q] = (unsigned int)pb[(16 + 2 * q) * HID] | ((unsigned int)pb[(17 + 2 * q) * HID] << 16);
      }
      acc = __builtin_amdgcn_wmma_f32_16x16x32_bf16(false, a.v, false, b.v,
                                                    (short)0, acc, false, false);
    }
    float* Dst = mat ? sK : sQ;
#pragma unroll
    for (int j = 0; j < 8; ++j)                 // C layout: VGPR j -> M=j(+8)
      Dst[(j + hh * 8) * LDF + ncol * 16 + ml] = acc[j];
  }
  __syncthreads();

  // per-row dot(Q,K): 16 threads x 16 elems, then 16-way LDS reduce
  {
    float p = 0.f;
    const float* qr = sQ + r * LDF + cb;
    const float* kr = sK + r * LDF + cb;
#pragma unroll
    for (int i = 0; i < 16; ++i) p += qr[i] * kr[i];
    sred[tid] = p;
  }
  __syncthreads();
  if (tid < 16) {
    float d = 0.f;
#pragma unroll
    for (int i = 0; i < 16; ++i) d += sred[tid * 16 + i];
    d *= SCALE;
    ws_dots[rowbase + tid] = d;
    const int seg = bidx[rowbase + tid];
    atomicMax(&segkey[seg], fkey(d));
  }
}

// ---------------------------------------------------------------------------
// Kernel 2: e = exp(dot - segmax); segment sum via float atomicAdd.
// ---------------------------------------------------------------------------
__global__ void k_softmax_part(const int* __restrict__ bidx,
                               const float* __restrict__ ws_dots,
                               const unsigned int* __restrict__ segkey,
                               float* __restrict__ ws_e, float* __restrict__ segsum) {
  int i = blockIdx.x * blockDim.x + threadIdx.x;
  int seg = bidx[i];
  float m = funkey(segkey[seg]);
  float e = __expf(ws_dots[i] - m);
  ws_e[i] = e;
  atomicAdd(&segsum[seg], e);
}

// ---------------------------------------------------------------------------
// Kernel 3: recompute V = drug@Wv+bv with WMMA (stays in LDS; saves a 256MB
// HBM round-trip), fuse attn*V + residual + LayerNorm, emit outputs.
// ---------------------------------------------------------------------------
__global__ __launch_bounds__(256)
void k_v_out(const float* __restrict__ drug, const int* __restrict__ bidx,
             const unsigned short* __restrict__ wvbf, const float* __restrict__ bv,
             const float* __restrict__ ws_e, const float* __restrict__ segsum,
             const float* __restrict__ gamma, const float* __restrict__ beta,
             float* __restrict__ out) {
  __shared__ unsigned short sAd[16 * LDA];
  __shared__ float sDr[16 * LDF];
  __shared__ float sV[16 * LDF];
  __shared__ float sred[256];
  __shared__ float sred2[256];
  __shared__ float sattn[16];
  __shared__ float smu[16];
  __shared__ float srs[16];

  const int tid = threadIdx.x;
  const int rowbase = blockIdx.x * 16;
  const int r  = tid >> 4;
  const int cb = (tid & 15) << 4;

  {
    const float4* dp = (const float4*)(drug + (size_t)rowbase * HID);
#pragma unroll
    for (int q = 0; q < 4; ++q) {
      float4 v = dp[tid * 4 + q];
      unsigned short* o = sAd + r * LDA + cb + q * 4;
      o[0] = f2bf(v.x); o[1] = f2bf(v.y); o[2] = f2bf(v.z); o[3] = f2bf(v.w);
      float* f = sDr + r * LDF + cb + q * 4;
      f[0] = v.x; f[1] = v.y; f[2] = v.z; f[3] = v.w;
    }
  }
  if (tid < 16) {
    int row = rowbase + tid;
    int seg = bidx[row];
    sattn[tid] = ws_e[row] / (segsum[seg] + SM_EPS);
  }
  __syncthreads();

  const int lane = tid & 31;
  const int wv   = tid >> 5;
  const int hh   = lane >> 4;
  const int ml   = lane & 15;

#pragma unroll
  for (int i = 0; i < 2; ++i) {               // 16 V column-tiles / 8 waves
    const int ncol = wv * 2 + i;
    const float bias = bv[ncol * 16 + ml];
    v8f acc;
#pragma unroll
    for (int j = 0; j < 8; ++j) acc[j] = bias;
#pragma unroll
    for (int k0 = 0; k0 < HID; k0 += 32) {
      BfPack a, b;
      const unsigned short* ar = sAd + ml * LDA + k0 + hh * 8;
      const unsigned int* p1 = (const unsigned int*)ar;
      const unsigned int* p2 = (const unsigned int*)(ar + 16);
#pragma unroll
      for (int q = 0; q < 4; ++q) { a.u[q] = p1[q]; a.u[4 + q] = p2[q]; }
      const unsigned short* pb =
          wvbf + (size_t)(k0 + hh * 8) * HID + ncol * 16 + ml;
#pragma unroll
      for (int q = 0; q < 4; ++q) {
        b.u[q] = (unsigned int)pb[(2 * q) * HID] | ((unsigned int)pb[(2 * q + 1) * HID] << 16);
        b.u[4 + q] = (unsigned int)pb[(16 + 2 * q) * HID] | ((unsigned int)pb[(17 + 2 * q) * HID] << 16);
      }
      acc = __builtin_amdgcn_wmma_f32_16x16x32_bf16(false, a.v, false, b.v,
                                                    (short)0, acc, false, false);
    }
#pragma unroll
    for (int j = 0; j < 8; ++j)
      sV[(j + hh * 8) * LDF + ncol * 16 + ml] = acc[j];
  }
  __syncthreads();

  // fused attn*V + residual, then LayerNorm stats (16 threads x 16 cols / row)
  const float a = sattn[r];
  float s = 0.f, s2 = 0.f;
#pragma unroll
  for (int i = 0; i < 16; ++i) {
    float o = a * sV[r * LDF + cb + i] + sDr[r * LDF + cb + i];
    s += o; s2 += o * o;
  }
  sred[tid] = s; sred2[tid] = s2;
  __syncthreads();
  if ((tid & 15) == 0) {
    float su = 0.f, sq = 0.f;
#pragma unroll
    for (int i = 0; i < 16; ++i) { su += sred[tid + i]; sq += sred2[tid + i]; }
    float mu  = su * (1.0f / HID);
    float var = sq * (1.0f / HID) - mu * mu;
    smu[r] = mu;
    srs[r] = rsqrtf(var + LN_EPS);
  }
  __syncthreads();
  const float mu = smu[r], rs = srs[r];
#pragma unroll
  for (int i = 0; i < 16; ++i) {
    int c = cb + i;
    float o = a * sV[r * LDF + c] + sDr[r * LDF + c];
    out[(size_t)(rowbase + r) * HID + c] = (o - mu) * rs * gamma[c] + beta[c];
  }
  if (tid < 16)
    out[(size_t)N_ATOMS * HID + rowbase + tid] = sattn[tid];  // attn output
}

// ---------------------------------------------------------------------------
extern "C" void kernel_launch(void* const* d_in, const int* in_sizes, int n_in,
                              void* d_out, int out_size, void* d_ws, size_t ws_size,
                              hipStream_t stream) {
  (void)in_sizes; (void)n_in; (void)out_size; (void)ws_size;
  const float* drug  = (const float*)d_in[0];
  const float* prot  = (const float*)d_in[1];
  const int*   bidx  = (const int*)d_in[2];
  const float* Wq    = (const float*)d_in[3];
  const float* bq    = (const float*)d_in[4];
  const float* Wk    = (const float*)d_in[5];
  const float* bk    = (const float*)d_in[6];
  const float* Wv    = (const float*)d_in[7];
  const float* bv    = (const float*)d_in[8];
  const float* gamma = (const float*)d_in[9];
  const float* beta  = (const float*)d_in[10];
  float* out = (float*)d_out;

  // Workspace layout (~2.5 MB total)
  float* ws = (float*)d_ws;
  float*        ws_dots = ws;                                   // N floats
  float*        ws_e    = ws + N_ATOMS;                         // N floats
  unsigned int* segkey  = (unsigned int*)(ws + 2 * (size_t)N_ATOMS);      // B
  float*        segsum  = ws + 2 * (size_t)N_ATOMS + BATCH;               // B
  unsigned short* wqbf  = (unsigned short*)(ws + 2 * (size_t)N_ATOMS + 2 * BATCH);
  unsigned short* wkbf  = wqbf + HID * HID;
  unsigned short* wvbf  = wkbf + HID * HID;

  k_init<<<(HID * HID) / 256, 256, 0, stream>>>(Wq, Wk, Wv, wqbf, wkbf, wvbf,
                                                segkey, segsum);
  k_qk_dots<<<N_ATOMS / 16, 256, 0, stream>>>(drug, prot, bidx, wqbf, wkbf,
                                              bq, bk, ws_dots, segkey);
  k_softmax_part<<<N_ATOMS / 256, 256, 0, stream>>>(bidx, ws_dots, segkey,
                                                    ws_e, segsum);
  k_v_out<<<N_ATOMS / 16, 256, 0, stream>>>(drug, bidx, wvbf, bv, ws_e, segsum,
                                            gamma, beta, out);
}
